// FLIP_20375324852288
// MI455X (gfx1250) — matrix operand order repaired and checked by reading, not compiled, not run on hardware
//
#include <hip/hip_runtime.h>
#include <math.h>

typedef __bf16 bf16_t;
typedef __attribute__((ext_vector_type(8)))  bf16_t v8bf;
typedef __attribute__((ext_vector_type(16))) bf16_t v16bf;
typedef __attribute__((ext_vector_type(8)))  float  v8f;

#define NLAYERS 6
#define DMODEL  1024
#define NHEADS  16
#define DHEAD   64
#define MLPDIM  4096
#define NPATCH  196
#define NKEEP   98
#define SEQ     99
#define NBATCH  32
#define ROWS    (NBATCH * SEQ)      /* 3168 */
#define PROWS   (NBATCH * NPATCH)   /* 6272 */
#define PKDIM   768

enum { GEMM_F32 = 0, GEMM_F32_POSEMB = 1, GEMM_F32_RESID = 2, GEMM_GELU_BF16 = 3 };

__device__ __forceinline__ float gelu_tanh(float x) {
  float x3 = x * x * x;
  return 0.5f * x * (1.0f + tanhf(0.7978845608028654f * (x + 0.044715f * x3)));
}

// ---------------------------------------------------------------------------
// Reformat a row-major f32 KxN weight matrix into bf16 WMMA B-fragment tiles.
// Tile (nb, kb) covers cols nb*16..+15, rows kb*32..+31. Within a tile each
// lane stores 16 contiguous bf16: lane -> n = lane&15, h = lane>>4,
// element e -> k = 8h + e (e<8), 16 + 8h + (e-8) (e>=8)  (matches A fragment order).
// Tiles ordered nb-major, kb-minor: streaming K reads contiguous 1KB chunks.
// ---------------------------------------------------------------------------
__global__ __launch_bounds__(256)
void prep_w_k(const float* __restrict__ W, bf16_t* __restrict__ Bf,
              int K, int N, int ntiles)
{
  int tile = blockIdx.x * 8 + (threadIdx.x >> 5);
  if (tile >= ntiles) return;
  int lane   = threadIdx.x & 31;
  int ktiles = K >> 5;
  int nb = tile / ktiles;
  int kb = tile - nb * ktiles;
  int nn = nb * 16 + (lane & 15);
  int kh = (lane >> 4) * 16;
  v16bf outv;
#pragma unroll
  for (int e = 0; e < 16; ++e) {
    int k = kb * 32 + (kh >> 1) + (e & 7) + ((e >> 3) << 4);
    outv[e] = (bf16_t)W[(size_t)k * N + nn];
  }
  *(v16bf*)(Bf + (size_t)tile * 512 + lane * 16) = outv;
}

// ---------------------------------------------------------------------------
// Patchify 32x224x224x3 f32 -> (6272 x 768) bf16 row-major activation matrix.
// ---------------------------------------------------------------------------
__global__ __launch_bounds__(256)
void patchify_k(const float* __restrict__ imgs, bf16_t* __restrict__ Ap)
{
  size_t i = (size_t)blockIdx.x * 256 + threadIdx.x;
  if (i >= (size_t)PROWS * PKDIM) return;
  int k   = (int)(i % PKDIM);
  int row = (int)(i / PKDIM);
  int n = row / NPATCH, l = row % NPATCH;
  int hi = l / 14, wj = l % 14;
  int c = k % 3, pp = k / 3;
  int pi = pp >> 4, pj = pp & 15;
  float v = imgs[(((size_t)n * 224 + hi * 16 + pi) * 224 + (wj * 16 + pj)) * 3 + c];
  Ap[i] = (bf16_t)v;
}

// ---------------------------------------------------------------------------
// WMMA bf16 GEMM: C[MxN] = A[MxK](bf16,row-major) @ Bfrag + bias, fused epilogue.
// 256 threads = 8 waves; wave tile 32x64: 2 A-frags x 4 B-frags -> 8 WMMAs per
// K-step on 12 b128 loads (each B frag feeds two WMMAs from registers).
// Block tile 32x512. Grid: (N/512, M/32). All dims exact -> EXEC all ones.
// A (HBM-streamed operand) is prefetched ~8 K-tiles ahead (global_prefetch_b8,
// speculative: past-the-end addresses are dropped per TH spec).
// ---------------------------------------------------------------------------
template <int MODE>
__global__ __launch_bounds__(256)
void gemm_wmma_k(const bf16_t* __restrict__ A, const bf16_t* __restrict__ Bf,
                 const float* __restrict__ bias, float* __restrict__ Cf,
                 bf16_t* __restrict__ Cb, const float* __restrict__ extra,
                 int M, int N, int K)
{
  const int lane = threadIdx.x & 31;
  const int wid  = threadIdx.x >> 5;
  const int m0 = blockIdx.y * 32;
  const int n0 = blockIdx.x * 512 + wid * 64;
  const int ktiles = K >> 5;

  // A fragment addressing: 16x32 bf16 A layout (ISA 7.12.2)
  const int kbase = (lane >> 4) << 3;
  const bf16_t* __restrict__ Arow0 = A + (size_t)(m0 + (lane & 15)) * K + kbase;
  const bf16_t* __restrict__ Arow1 = Arow0 + (size_t)16 * K;

  const bf16_t* __restrict__ Bp[4];
#pragma unroll
  for (int j = 0; j < 4; ++j)
    Bp[j] = Bf + (size_t)((n0 >> 4) + j) * ktiles * 512 + lane * 16;

  v8f acc0[4] = {};
  v8f acc1[4] = {};

  for (int kt = 0; kt < ktiles; ++kt) {
    const bf16_t* a0 = Arow0 + (kt << 5);
    const bf16_t* a1 = Arow1 + (kt << 5);
    v16bf af0 = __builtin_shufflevector(*(const v8bf*)a0, *(const v8bf*)(a0 + 16),
                  0,1,2,3,4,5,6,7,8,9,10,11,12,13,14,15);
    v16bf af1 = __builtin_shufflevector(*(const v8bf*)a1, *(const v8bf*)(a1 + 16),
                  0,1,2,3,4,5,6,7,8,9,10,11,12,13,14,15);
    __builtin_prefetch(a0 + 256, 0, 0);   // 8 K-tiles ahead
    __builtin_prefetch(a1 + 256, 0, 0);
#pragma unroll
    for (int j = 0; j < 4; ++j) {
      v16bf b = *(const v16bf*)(Bp[j] + (size_t)kt * 512);
      acc0[j] = __builtin_amdgcn_wmma_f32_16x16x32_bf16(false, af0, false, b, (short)0, acc0[j], false, false);
      acc1[j] = __builtin_amdgcn_wmma_f32_16x16x32_bf16(false, af1, false, b, (short)0, acc1[j], false, false);
    }
  }

  // C/D layout: VGPR r -> row +r + 8*(lane>>4); col = n0 + 16j + (lane&15)
  const int nlo = lane & 15;
  const int rb  = (lane >> 4) << 3;
#pragma unroll
  for (int j = 0; j < 4; ++j) {
    const int col = n0 + j * 16 + nlo;
    const float bv = bias[col];
#pragma unroll
    for (int r = 0; r < 8; ++r) {
#pragma unroll
      for (int half = 0; half < 2; ++half) {
        const int row = m0 + half * 16 + rb + r;
        const size_t idx = (size_t)row * N + col;
        float v = (half ? acc1[j][r] : acc0[j][r]) + bv;
        if constexpr (MODE == GEMM_F32) {
          Cf[idx] = v;
        } else if constexpr (MODE == GEMM_F32_POSEMB) {
          Cf[idx] = v + extra[(size_t)(row % NPATCH) * N + col];
        } else if constexpr (MODE == GEMM_F32_RESID) {
          Cf[idx] = Cf[idx] + v;
        } else {
          Cb[idx] = (bf16_t)gelu_tanh(v);
        }
      }
    }
  }
}

// ---------------------------------------------------------------------------
// LayerNorm over D=1024; one block per row. BF16OUT -> bf16 (GEMM A buffer),
// else f32 (final LN into d_out).
// ---------------------------------------------------------------------------
template <bool BF16OUT>
__global__ __launch_bounds__(256)
void layernorm_k(const float* __restrict__ X, const float* __restrict__ gs,
                 const float* __restrict__ gb, bf16_t* __restrict__ Yb,
                 float* __restrict__ Yf)
{
  __shared__ float red[256];
  const int row = blockIdx.x;
  const int tid = threadIdx.x;
  const float* xr = X + (size_t)row * DMODEL;
  float v0 = xr[tid], v1 = xr[tid + 256], v2 = xr[tid + 512], v3 = xr[tid + 768];
  red[tid] = v0 + v1 + v2 + v3;
  __syncthreads();
  for (int s = 128; s > 0; s >>= 1) { if (tid < s) red[tid] += red[tid + s]; __syncthreads(); }
  const float mean = red[0] * (1.0f / DMODEL);
  __syncthreads();
  float d0 = v0 - mean, d1 = v1 - mean, d2 = v2 - mean, d3 = v3 - mean;
  red[tid] = d0 * d0 + d1 * d1 + d2 * d2 + d3 * d3;
  __syncthreads();
  for (int s = 128; s > 0; s >>= 1) { if (tid < s) red[tid] += red[tid + s]; __syncthreads(); }
  const float rstd = rsqrtf(red[0] * (1.0f / DMODEL) + 1e-6f);
  float dv[4] = {d0, d1, d2, d3};
#pragma unroll
  for (int k = 0; k < 4; ++k) {
    int c = tid + k * 256;
    float y = dv[k] * rstd * gs[c] + gb[c];
    if constexpr (BF16OUT) Yb[(size_t)row * DMODEL + c] = (bf16_t)y;
    else                   Yf[(size_t)row * DMODEL + c] = y;
  }
}

// ---------------------------------------------------------------------------
// Masking: per-image stable argsort ranks via comparison count; writes mask
// to d_out tail and gathers [cls | kept patches] into the f32 sequence buffer.
// ---------------------------------------------------------------------------
__global__ __launch_bounds__(256)
void mask_gather_k(const float* __restrict__ noise, const float* __restrict__ x0,
                   const float* __restrict__ cls, float* __restrict__ Xs,
                   float* __restrict__ mask_out)
{
  __shared__ float ns[NPATCH];
  __shared__ int keep[NKEEP];
  const int n = blockIdx.x;
  for (int i = threadIdx.x; i < NPATCH; i += 256) ns[i] = noise[(size_t)n * NPATCH + i];
  __syncthreads();
  if (threadIdx.x < NPATCH) {
    const int t = threadIdx.x;
    const float vt = ns[t];
    int rank = 0;
    for (int j = 0; j < NPATCH; ++j) {
      float vj = ns[j];
      rank += (vj < vt) || (vj == vt && j < t);   // stable sort semantics
    }
    mask_out[(size_t)n * NPATCH + t] = (rank < NKEEP) ? 0.0f : 1.0f;
    if (rank < NKEEP) keep[rank] = t;
  }
  __syncthreads();
  for (int i = threadIdx.x; i < SEQ * DMODEL; i += 256) {
    int s = i >> 10, c = i & (DMODEL - 1);
    float v = (s == 0) ? cls[c] : x0[((size_t)n * NPATCH + keep[s - 1]) * DMODEL + c];
    Xs[((size_t)n * SEQ + s) * DMODEL + c] = v;
  }
}

// ---------------------------------------------------------------------------
// Attention (3% of FLOPs -> VALU flash-style). Block per (n,h); K/V head tile
// staged in LDS (2 * 99*64*4B = 50.7KB of the 320KB WGP LDS); one query row
// per thread with online softmax; writes bf16 o (A-operand of the wo GEMM).
// ---------------------------------------------------------------------------
__global__ __launch_bounds__(128)
void attention_k(const float* __restrict__ Q, const float* __restrict__ Kb,
                 const float* __restrict__ Vb, bf16_t* __restrict__ Ob)
{
  __shared__ float Ks[SEQ * DHEAD];
  __shared__ float Vs[SEQ * DHEAD];
  const int n = blockIdx.x;
  const int h = blockIdx.y;
  const size_t base = (size_t)n * SEQ * DMODEL + (size_t)h * DHEAD;
  for (int i = threadIdx.x; i < SEQ * DHEAD; i += 128) {
    int s = i >> 6, d = i & 63;
    Ks[i] = Kb[base + (size_t)s * DMODEL + d];
    Vs[i] = Vb[base + (size_t)s * DMODEL + d];
  }
  __syncthreads();
  const int qi = threadIdx.x;
  if (qi < SEQ) {
    float q[DHEAD];
    const float* qr = Q + base + (size_t)qi * DMODEL;
#pragma unroll
    for (int d = 0; d < DHEAD; ++d) q[d] = qr[d];
    float mx = -3.0e38f, l = 0.0f;
    float acc[DHEAD];
#pragma unroll
    for (int d = 0; d < DHEAD; ++d) acc[d] = 0.0f;
    for (int kk = 0; kk < SEQ; ++kk) {
      const float* kr = &Ks[kk * DHEAD];
      float s = 0.0f;
#pragma unroll
      for (int d = 0; d < DHEAD; ++d) s += q[d] * kr[d];
      s *= 0.125f;                       // 1/sqrt(64)
      float mn   = fmaxf(mx, s);
      float corr = __expf(mx - mn);
      float p    = __expf(s - mn);
      l = l * corr + p;
      const float* vr = &Vs[kk * DHEAD];
#pragma unroll
      for (int d = 0; d < DHEAD; ++d) acc[d] = acc[d] * corr + p * vr[d];
      mx = mn;
    }
    float inv = 1.0f / l;
    bf16_t* orow = Ob + base + (size_t)qi * DMODEL;
#pragma unroll
    for (int d = 0; d < DHEAD; ++d) orow[d] = (bf16_t)(acc[d] * inv);
  }
}

// ---------------------------------------------------------------------------
extern "C" void kernel_launch(void* const* d_in, const int* in_sizes, int n_in,
                              void* d_out, int out_size, void* d_ws, size_t ws_size,
                              hipStream_t stream) {
  (void)in_sizes; (void)n_in; (void)out_size; (void)ws_size;
  const float* imgs    = (const float*)d_in[0];
  const float* noise   = (const float*)d_in[1];
  const float* patch_w = (const float*)d_in[2];
  const float* patch_b = (const float*)d_in[3];
  const float* pos_emb = (const float*)d_in[4];
  const float* cls_tok = (const float*)d_in[5];
  const float* ln1_s   = (const float*)d_in[6];
  const float* ln1_b   = (const float*)d_in[7];
  const float* wq      = (const float*)d_in[8];
  const float* bq      = (const float*)d_in[9];
  const float* wk      = (const float*)d_in[10];
  const float* bk      = (const float*)d_in[11];
  const float* wv      = (const float*)d_in[12];
  const float* bv      = (const float*)d_in[13];
  const float* wo      = (const float*)d_in[14];
  const float* bo      = (const float*)d_in[15];
  const float* ln2_s   = (const float*)d_in[16];
  const float* ln2_b   = (const float*)d_in[17];
  const float* w1      = (const float*)d_in[18];
  const float* b1      = (const float*)d_in[19];
  const float* w2      = (const float*)d_in[20];
  const float* b2      = (const float*)d_in[21];
  const float* lnf_s   = (const float*)d_in[22];
  const float* lnf_b   = (const float*)d_in[23];
  float* out = (float*)d_out;

  // ---- workspace layout (256B aligned) ----
  size_t off = 0;
  char* wsb = (char*)d_ws;
  auto take = [&](size_t bytes) -> char* {
    char* p = wsb + off;
    off = (off + bytes + 255) & ~(size_t)255;
    return p;
  };
  const size_t LAYER_FRAG = 4ull * 1024 * 1024 + 2ull * 1024 * 4096; // 12,582,912 bf16
  bf16_t* wf_patch = (bf16_t*)take((size_t)PKDIM * DMODEL * 2);
  bf16_t* wf_layer = (bf16_t*)take((size_t)NLAYERS * LAYER_FRAG * 2);
  bf16_t* apatch   = (bf16_t*)take((size_t)PROWS * PKDIM * 2);
  float*  x0       = (float*)take((size_t)PROWS * DMODEL * 4);
  float*  xs       = (float*)take((size_t)ROWS * DMODEL * 4);
  bf16_t* yb       = (bf16_t*)take((size_t)ROWS * DMODEL * 2);
  float*  qbuf     = (float*)take((size_t)ROWS * DMODEL * 4);
  float*  kbuf     = (float*)take((size_t)ROWS * DMODEL * 4);
  float*  vbuf     = (float*)take((size_t)ROWS * DMODEL * 4);
  bf16_t* obuf     = (bf16_t*)take((size_t)ROWS * DMODEL * 2);
  bf16_t* hbuf     = (bf16_t*)take((size_t)ROWS * MLPDIM * 2);

  // ---- reformat all weights to bf16 WMMA fragment tiles (L2-resident) ----
  prep_w_k<<<dim3((64 * 24) / 8), 256, 0, stream>>>(patch_w, wf_patch, PKDIM, DMODEL, 64 * 24);
  for (int i = 0; i < NLAYERS; ++i) {
    bf16_t* wf = wf_layer + (size_t)i * LAYER_FRAG;
    const size_t dd = (size_t)DMODEL * DMODEL;
    prep_w_k<<<dim3(2048 / 8), 256, 0, stream>>>(wq + i * dd, wf,           DMODEL, DMODEL, 2048);
    prep_w_k<<<dim3(2048 / 8), 256, 0, stream>>>(wk + i * dd, wf + 1048576, DMODEL, DMODEL, 2048);
    prep_w_k<<<dim3(2048 / 8), 256, 0, stream>>>(wv + i * dd, wf + 2097152, DMODEL, DMODEL, 2048);
    prep_w_k<<<dim3(2048 / 8), 256, 0, stream>>>(wo + i * dd, wf + 3145728, DMODEL, DMODEL, 2048);
    prep_w_k<<<dim3(8192 / 8), 256, 0, stream>>>(w1 + (size_t)i * DMODEL * MLPDIM, wf + 4194304, DMODEL, MLPDIM, 8192);
    prep_w_k<<<dim3(8192 / 8), 256, 0, stream>>>(w2 + (size_t)i * MLPDIM * DMODEL, wf + 8388608, MLPDIM, DMODEL, 8192);
  }

  // ---- patch embed: patchify -> WMMA GEMM (+bias +pos_emb) ----
  patchify_k<<<dim3(((size_t)PROWS * PKDIM) / 256), 256, 0, stream>>>(imgs, apatch);
  gemm_wmma_k<GEMM_F32_POSEMB><<<dim3(DMODEL / 512, PROWS / 32), 256, 0, stream>>>(
      apatch, wf_patch, patch_b, x0, nullptr, pos_emb, PROWS, DMODEL, PKDIM);

  // ---- masking + gather + cls concat; mask -> tail of d_out ----
  float* mask_out = out + (size_t)NBATCH * SEQ * DMODEL;
  mask_gather_k<<<dim3(NBATCH), 256, 0, stream>>>(noise, x0, cls_tok, xs, mask_out);

  // ---- transformer layers ----
  const dim3 g1024(DMODEL / 512, ROWS / 32);   // (2, 99)
  const dim3 g4096(MLPDIM / 512, ROWS / 32);   // (8, 99)
  for (int i = 0; i < NLAYERS; ++i) {
    bf16_t* wf = wf_layer + (size_t)i * LAYER_FRAG;
    layernorm_k<true><<<dim3(ROWS), 256, 0, stream>>>(xs, ln1_s + i * DMODEL, ln1_b + i * DMODEL, yb, nullptr);
    gemm_wmma_k<GEMM_F32><<<g1024, 256, 0, stream>>>(yb, wf,           bq + i * DMODEL, qbuf, nullptr, nullptr, ROWS, DMODEL, DMODEL);
    gemm_wmma_k<GEMM_F32><<<g1024, 256, 0, stream>>>(yb, wf + 1048576, bk + i * DMODEL, kbuf, nullptr, nullptr, ROWS, DMODEL, DMODEL);
    gemm_wmma_k<GEMM_F32><<<g1024, 256, 0, stream>>>(yb, wf + 2097152, bv + i * DMODEL, vbuf, nullptr, nullptr, ROWS, DMODEL, DMODEL);
    attention_k<<<dim3(NBATCH, NHEADS), 128, 0, stream>>>(qbuf, kbuf, vbuf, obuf);
    gemm_wmma_k<GEMM_F32_RESID><<<g1024, 256, 0, stream>>>(obuf, wf + 3145728, bo + i * DMODEL, xs, nullptr, nullptr, ROWS, DMODEL, DMODEL);
    layernorm_k<true><<<dim3(ROWS), 256, 0, stream>>>(xs, ln2_s + i * DMODEL, ln2_b + i * DMODEL, yb, nullptr);
    gemm_wmma_k<GEMM_GELU_BF16><<<g4096, 256, 0, stream>>>(yb, wf + 4194304, b1 + i * MLPDIM, nullptr, hbuf, nullptr, ROWS, MLPDIM, DMODEL);
    gemm_wmma_k<GEMM_F32_RESID><<<g1024, 256, 0, stream>>>(hbuf, wf + 8388608, b2 + i * DMODEL, xs, nullptr, nullptr, ROWS, DMODEL, MLPDIM);
  }

  // ---- final LN -> x part of d_out ----
  layernorm_k<false><<<dim3(ROWS), 256, 0, stream>>>(xs, lnf_s, lnf_b, nullptr, out);
}
